// BidirAutoregMamba_67662914781229
// MI455X (gfx1250) — matrix operand order repaired and checked by reading, not compiled, not run on hardware
//
#include <hip/hip_runtime.h>
#include <hip/hip_bf16.h>
#include <math.h>

// ---- model dims (from reference) ----
#define L_SEQ   1026      // SEQ + sos + eos
#define DMODEL  1024
#define DINNER  2048
#define DSTATE  64
#define DTRANK  64
#define NXPROJ  192       // DT_RANK + 2*D_STATE
#define NHID    2048

typedef __attribute__((ext_vector_type(16))) __bf16 v16bf_t;
typedef __attribute__((ext_vector_type(8)))  float  v8f_t;
union FragBF { v16bf_t v; unsigned int u[8]; };

__device__ __forceinline__ unsigned short f32_to_bf16_rn(float f) {
  unsigned int x = __float_as_uint(f);
  unsigned int r = x + 0x7FFFu + ((x >> 16) & 1u);   // round-to-nearest-even
  return (unsigned short)(r >> 16);
}
__device__ __forceinline__ float silu_f(float x)     { return x / (1.0f + __expf(-x)); }
__device__ __forceinline__ float softplus_f(float x) { return x > 20.0f ? x : log1pf(__expf(x)); }

// gfx1250 async global->LDS copy (ASYNCcnt tracked). VDST = per-lane LDS byte
// offset (generic shared pointers carry the LDS offset in addr[31:0]).
__device__ __forceinline__ void async_copy_b128(unsigned lds_off, const unsigned short* g) {
  asm volatile("global_load_async_to_lds_b128 %0, %1, off"
               :: "v"(lds_off), "v"((unsigned long long)(size_t)g)
               : "memory");
}

__device__ __forceinline__ v8f_t wmma_bf16(const FragBF& a, const FragBF& b, v8f_t c) {
  return __builtin_amdgcn_wmma_f32_16x16x32_bf16(false, a.v, false, b.v,
                                                 (short)0, c, false, false);
}

// ---------------- weight convert f32[K][N] -> bf16 transposed [N][K] ----------------
__global__ __launch_bounds__(256) void cvt_bf16_t_kernel(const float* __restrict__ src,
    unsigned short* __restrict__ dst, int K, int N) {
  int i = blockIdx.x * 256 + threadIdx.x;
  if (i >= K * N) return;
  int k = i / N, n = i - k * N;
  dst[(size_t)n * K + k] = f32_to_bf16_rn(src[i]);
}

// ---------------- [sos; hidden; eos] ----------------
__global__ __launch_bounds__(256) void embed_kernel(const float* __restrict__ hid,
    const float* __restrict__ sos, const float* __restrict__ eos, float* __restrict__ h) {
  int i = blockIdx.x * 256 + threadIdx.x;
  if (i >= L_SEQ * DMODEL) return;
  int t = i >> 10, d = i & (DMODEL - 1);
  h[i] = (t == 0) ? sos[d] : (t == L_SEQ - 1) ? eos[d] : hid[(size_t)(t - 1) * DMODEL + d];
}

__global__ __launch_bounds__(256) void add_kernel(const float* __restrict__ a,
    const float* __restrict__ b, float* __restrict__ c, int n) {
  int i = blockIdx.x * 256 + threadIdx.x;
  if (i < n) c[i] = a[i] + b[i];
}

__global__ __launch_bounds__(256) void flip_kernel(const float* __restrict__ src,
                                                   float* __restrict__ dst) {
  int i = blockIdx.x * 256 + threadIdx.x;
  if (i >= L_SEQ * DMODEL) return;
  int t = i >> 10, d = i & (DMODEL - 1);
  dst[i] = src[(size_t)(L_SEQ - 1 - t) * DMODEL + d];
}

// ---------------- rmsnorm (block per row, D=1024) -> bf16 for next GEMM ----------------
__global__ __launch_bounds__(256) void rmsnorm_bf_kernel(const float* __restrict__ x,
    const float* __restrict__ w, unsigned short* __restrict__ y) {
  const int row = blockIdx.x;
  const float* xr = x + (size_t)row * DMODEL;
  __shared__ float red[256];
  float s = 0.f;
  for (int i = threadIdx.x; i < DMODEL; i += 256) { float v = xr[i]; s += v * v; }
  red[threadIdx.x] = s; __syncthreads();
  for (int st = 128; st > 0; st >>= 1) {
    if (threadIdx.x < st) red[threadIdx.x] += red[threadIdx.x + st];
    __syncthreads();
  }
  const float scale = rsqrtf(red[0] / (float)DMODEL + 1e-5f);
  for (int i = threadIdx.x; i < DMODEL; i += 256)
    y[(size_t)row * DMODEL + i] = f32_to_bf16_rn(xr[i] * scale * w[i]);
}

__global__ __launch_bounds__(256) void layernorm_kernel(const float* __restrict__ x,
    const float* __restrict__ w, const float* __restrict__ b, float* __restrict__ y) {
  const int row = blockIdx.x;
  const float* xr = x + (size_t)row * DMODEL;
  __shared__ float red[256];
  float s = 0.f;
  for (int i = threadIdx.x; i < DMODEL; i += 256) s += xr[i];
  red[threadIdx.x] = s; __syncthreads();
  for (int st = 128; st > 0; st >>= 1) {
    if (threadIdx.x < st) red[threadIdx.x] += red[threadIdx.x + st];
    __syncthreads();
  }
  const float mean = red[0] / (float)DMODEL;
  __syncthreads();
  float v = 0.f;
  for (int i = threadIdx.x; i < DMODEL; i += 256) { float d = xr[i] - mean; v += d * d; }
  red[threadIdx.x] = v; __syncthreads();
  for (int st = 128; st > 0; st >>= 1) {
    if (threadIdx.x < st) red[threadIdx.x] += red[threadIdx.x + st];
    __syncthreads();
  }
  const float inv = rsqrtf(red[0] / (float)DMODEL + 1e-5f);
  for (int i = threadIdx.x; i < DMODEL; i += 256)
    y[(size_t)row * DMODEL + i] = (xr[i] - mean) * inv * w[i] + b[i];
}

// ---------------- depthwise causal conv (K=4) + silu -> f32 (scan) + bf16 (GEMM) ----------------
__global__ __launch_bounds__(256) void conv_silu_kernel(const float* __restrict__ xz,
    const float* __restrict__ w, const float* __restrict__ b,
    float* __restrict__ out, unsigned short* __restrict__ out_bf, int n) {
  int i = blockIdx.x * 256 + threadIdx.x;
  if (i >= n) return;
  int t = i / DINNER, d = i - t * DINNER;
  float acc = b[d];
  #pragma unroll
  for (int j = 0; j < 4; ++j) {
    int tt = t - 3 + j;
    if (tt >= 0) acc += w[d * 4 + j] * xz[(size_t)tt * (2 * DINNER) + d];
  }
  float v = silu_f(acc);
  out[i] = v;
  out_bf[i] = f32_to_bf16_rn(v);
}

// y * silu(z) -> bf16 (feeds out_proj GEMM)
__global__ __launch_bounds__(256) void gate_mul_kernel(const float* __restrict__ y,
    const float* __restrict__ xz, unsigned short* __restrict__ out, int n) {
  int i = blockIdx.x * 256 + threadIdx.x;
  if (i >= n) return;
  int t = i / DINNER, d = i - t * DINNER;
  out[i] = f32_to_bf16_rn(y[i] * silu_f(xz[(size_t)t * (2 * DINNER) + DINNER + d]));
}

// GatedMLP combine: y * silu(gate) -> bf16 (feeds fc2 GEMM)
__global__ __launch_bounds__(256) void glu_kernel(const float* __restrict__ fc,
                                                  unsigned short* __restrict__ out, int n) {
  int i = blockIdx.x * 256 + threadIdx.x;
  if (i >= n) return;
  int t = i / NHID, d = i - t * NHID;
  const float* row = fc + (size_t)t * (2 * NHID);
  out[i] = f32_to_bf16_rn(row[d] * silu_f(row[NHID + d]));
}

// ---------------- selective scan: one wave32 per channel, 2 states/lane ----------------
__global__ __launch_bounds__(256) void scan_kernel(
    const float* __restrict__ u, const float* __restrict__ dt,
    const float* __restrict__ A_log, const float* __restrict__ dbc,
    const float* __restrict__ Dv, float* __restrict__ y)
{
  const int wave = threadIdx.x >> 5;
  const int lane = threadIdx.x & 31;
  const int d = blockIdx.x * 8 + wave;
  const float A0 = -__expf(A_log[(size_t)d * DSTATE + 2 * lane]);
  const float A1 = -__expf(A_log[(size_t)d * DSTATE + 2 * lane + 1]);
  const float Dd = Dv[d];
  float h0 = 0.f, h1 = 0.f;
  for (int t = 0; t < L_SEQ; ++t) {
    const float ut  = u [(size_t)t * DINNER + d];
    const float dtt = dt[(size_t)t * DINNER + d];
    const float* bc = dbc + (size_t)t * NXPROJ;
    const float B0 = bc[DTRANK + 2 * lane];
    const float B1 = bc[DTRANK + 2 * lane + 1];
    const float C0 = bc[DTRANK + DSTATE + 2 * lane];
    const float C1 = bc[DTRANK + DSTATE + 2 * lane + 1];
    const float du = dtt * ut;
    h0 = h0 * __expf(dtt * A0) + du * B0;
    h1 = h1 * __expf(dtt * A1) + du * B1;
    float p = h0 * C0 + h1 * C1;
    #pragma unroll
    for (int m = 16; m >= 1; m >>= 1) p += __shfl_xor(p, m, 32);   // wave32 reduce
    if (lane == 0) y[(size_t)t * DINNER + d] = p + ut * Dd;
  }
}

// ---------------- WMMA GEMM: C = epi(Abf[M,K] @ Wt[N,K]^T + bias) ----------------
// Block tile 128x64, K-step 32, 8 waves (4x2), each wave 32x32 = 4 WMMA / step.
// Double-buffered LDS staged with global_load_async_to_lds_b128 (ASYNCcnt).
// mode: 0 = none, 1 = +bias, 2 = softplus(+bias). Optional bf16 mirror Cbf.
#define LDSK 56   // padded k-stride (ushorts): 112B = 16B-aligned, conflict-free gathers
__global__ __launch_bounds__(256) void wmma_gemm_kernel(
    const unsigned short* __restrict__ Abf, int lda,
    const unsigned short* __restrict__ Wt,       // [N][K] bf16
    const float* __restrict__ bias,
    float* __restrict__ C, unsigned short* __restrict__ Cbf, int ldc,
    int M, int N, int K, int mode)
{
  __shared__ __align__(16) unsigned short As[2][128][LDSK];
  __shared__ __align__(16) unsigned short Bs[2][64][LDSK];

  const int tid   = threadIdx.x;
  const int lane  = tid & 31;
  const int wave  = tid >> 5;
  const int waveM = wave >> 1;            // 0..3 -> rows 0/32/64/96
  const int waveN = wave & 1;             // 0..1 -> cols 0/32
  const int mBase = blockIdx.y * 128;
  const int nBase = blockIdx.x * 64;
  const int lh = lane >> 4;               // lane half
  const int ll = lane & 15;

  // staging: each thread copies 3x16B per K-tile (2 for A, 1 for B)
  const int sRow = tid >> 2;              // 0..63
  const int sKc  = (tid & 3) * 8;         // 0,8,16,24 (bf16 elems)
  const int am0  = (mBase + sRow      < M) ? (mBase + sRow)      : (M - 1); // clamp: rows >=M unused
  const int am1  = (mBase + sRow + 64 < M) ? (mBase + sRow + 64) : (M - 1);
  const int bn   = nBase + sRow;

  v8f_t acc00 = {}, acc01 = {}, acc10 = {}, acc11 = {};

  auto issue_tile = [&](int buf, int k0) {
    async_copy_b128((unsigned)(size_t)&As[buf][sRow     ][sKc], Abf + (size_t)am0 * lda + k0 + sKc);
    async_copy_b128((unsigned)(size_t)&As[buf][sRow + 64][sKc], Abf + (size_t)am1 * lda + k0 + sKc);
    async_copy_b128((unsigned)(size_t)&Bs[buf][sRow     ][sKc], Wt  + (size_t)bn  * K   + k0 + sKc);
  };

  const int nK = K >> 5;
  issue_tile(0, 0);
  for (int kt = 0; kt < nK; ++kt) {
    if (kt + 1 < nK) {
      issue_tile((kt + 1) & 1, (kt + 1) << 5);
      asm volatile("s_wait_asynccnt 0x3" ::: "memory");   // current tile's 3 copies done
    } else {
      asm volatile("s_wait_asynccnt 0x0" ::: "memory");
    }
    __syncthreads();

    if (kt + 2 < nK)   // global_prefetch_b8 on the tile after next
      __builtin_prefetch(Wt + (size_t)bn * K + ((kt + 2) << 5) + sKc, 0, 1);

    const int buf = kt & 1;
    // fragments per CDNA5 16-bit WMMA VGPR layouts
    FragBF a0, a1, b0, b1;
    #pragma unroll
    for (int r = 0; r < 8; ++r) {
      const int kA = ((r & 4) << 2) + lh * 8 + (r & 3) * 2;   // A: {0..7,16..23}/{8..15,24..31}
      const int kB = ((r & 4) << 2) + lh * 8 + (r & 3) * 2;   // B: {0..7,16..23}/{8..15,24..31}
      a0.u[r] = *(const unsigned int*)&As[buf][waveM * 32 +      ll][kA];
      a1.u[r] = *(const unsigned int*)&As[buf][waveM * 32 + 16 + ll][kA];
      b0.u[r] = *(const unsigned int*)&Bs[buf][waveN * 32 +      ll][kB];
      b1.u[r] = *(const unsigned int*)&Bs[buf][waveN * 32 + 16 + ll][kB];
    }
    acc00 = wmma_bf16(a0, b0, acc00);
    acc01 = wmma_bf16(a0, b1, acc01);
    acc10 = wmma_bf16(a1, b0, acc10);
    acc11 = wmma_bf16(a1, b1, acc11);
    __syncthreads();
  }

  // epilogue: C/D layout -> VGPR r holds (m = r + 8*lh, n = ll)
  const int n0 = nBase + waveN * 32 + ll;
  const int n1 = n0 + 16;
  const float bv0 = (mode >= 1) ? bias[n0] : 0.0f;
  const float bv1 = (mode >= 1) ? bias[n1] : 0.0f;
  #pragma unroll
  for (int r = 0; r < 8; ++r) {
    const int m0 = mBase + waveM * 32 + r + 8 * lh;
    const int m1 = m0 + 16;
    float v00 = acc00[r] + bv0, v01 = acc01[r] + bv1;
    float v10 = acc10[r] + bv0, v11 = acc11[r] + bv1;
    if (mode == 2) {
      v00 = softplus_f(v00); v01 = softplus_f(v01);
      v10 = softplus_f(v10); v11 = softplus_f(v11);
    }
    if (m0 < M) {
      C[(size_t)m0 * ldc + n0] = v00;
      C[(size_t)m0 * ldc + n1] = v01;
      if (Cbf) { Cbf[(size_t)m0 * ldc + n0] = f32_to_bf16_rn(v00);
                 Cbf[(size_t)m0 * ldc + n1] = f32_to_bf16_rn(v01); }
    }
    if (m1 < M) {
      C[(size_t)m1 * ldc + n0] = v10;
      C[(size_t)m1 * ldc + n1] = v11;
      if (Cbf) { Cbf[(size_t)m1 * ldc + n0] = f32_to_bf16_rn(v10);
                 Cbf[(size_t)m1 * ldc + n1] = f32_to_bf16_rn(v11); }
    }
  }
}

static inline int ceil_div(int a, int b) { return (a + b - 1) / b; }

extern "C" void kernel_launch(void* const* d_in, const int* in_sizes, int n_in,
                              void* d_out, int out_size, void* d_ws, size_t ws_size,
                              hipStream_t stream) {
  (void)in_sizes; (void)n_in; (void)out_size; (void)ws_size;
  const float* hidden = (const float*)d_in[0];
  const float* sos    = (const float*)d_in[1];
  const float* eos    = (const float*)d_in[2];
  const float* lnw    = (const float*)d_in[3];
  const float* lnb    = (const float*)d_in[4];

  // per-layer param order: 0 in_proj, 1 conv_w, 2 conv_b, 3 x_proj, 4 dt_w, 5 dt_b,
  // 6 A_log, 7 D, 8 out_proj, 9 norm1_w, 10 norm2_w, 11 fc1_w, 12 fc1_b, 13 fc2_w, 14 fc2_b
  const float* Wp[2][15];
  for (int l = 0; l < 2; ++l)
    for (int j = 0; j < 15; ++j)
      Wp[l][j] = (const float*)d_in[5 + l * 15 + j];

  // workspace bump allocator (256B aligned)
  char* p = (char*)d_ws;
  auto allocf = [&](size_t n) { float* r = (float*)p;
    p += ((n * sizeof(float) + 255) & ~(size_t)255); return r; };
  auto allocu = [&](size_t n) { unsigned short* r = (unsigned short*)p;
    p += ((n * sizeof(unsigned short) + 255) & ~(size_t)255); return r; };

  const size_t LD = (size_t)L_SEQ * DMODEL;
  float* h       = allocf(LD);
  float* res     = allocf(LD);
  float* hflip   = allocf(LD);
  float* resflip = allocf(LD);
  float* xz      = allocf((size_t)L_SEQ * 2 * DINNER);
  float* xconv   = allocf((size_t)L_SEQ * DINNER);
  float* dbc     = allocf((size_t)L_SEQ * NXPROJ);
  float* dtb     = allocf((size_t)L_SEQ * DINNER);
  float* yscan   = allocf((size_t)L_SEQ * DINNER);
  float* mout    = allocf(LD);
  float* fcout   = allocf((size_t)L_SEQ * 2 * NHID);
  float* ntp     = allocf(LD);
  float* ptp     = allocf(LD);

  unsigned short* xnorm_bf  = allocu(LD);
  unsigned short* xconv_bf  = allocu((size_t)L_SEQ * DINNER);
  unsigned short* dbc_bf    = allocu((size_t)L_SEQ * NXPROJ);
  unsigned short* ygated_bf = allocu((size_t)L_SEQ * DINNER);
  unsigned short* h2n_bf    = allocu(LD);
  unsigned short* glub_bf   = allocu((size_t)L_SEQ * NHID);

  // transposed bf16 weights Wt[N][K]: in_proj, x_proj, dt_w, out_proj, fc1_w, fc2_w
  unsigned short* wt[2][6];
  const int wK[6] = { DMODEL, DINNER, DTRANK, DINNER, DMODEL, NHID };
  const int wN[6] = { 2 * DINNER, NXPROJ, DINNER, DMODEL, 2 * NHID, DMODEL };
  const int widx[6] = {0, 3, 4, 8, 11, 13};
  for (int l = 0; l < 2; ++l)
    for (int j = 0; j < 6; ++j) {
      const int n = wK[j] * wN[j];
      wt[l][j] = allocu((size_t)n);
      cvt_bf16_t_kernel<<<ceil_div(n, 256), 256, 0, stream>>>(Wp[l][widx[j]], wt[l][j], wK[j], wN[j]);
    }

  embed_kernel<<<ceil_div((int)LD, 256), 256, 0, stream>>>(hidden, sos, eos, h);
  hipMemcpyAsync(res, h, LD * sizeof(float), hipMemcpyDeviceToDevice, stream); // res = h (layer 0)

  auto gemm = [&](const unsigned short* Abf, int lda, const unsigned short* Wtp,
                  const float* bias, float* Cout, unsigned short* Cbf,
                  int N, int K, int mode) {
    dim3 grid(N / 64, ceil_div(L_SEQ, 128));
    wmma_gemm_kernel<<<grid, 256, 0, stream>>>(Abf, lda, Wtp, bias, Cout, Cbf, N,
                                               L_SEQ, N, K, mode);
  };

  const int nEl = L_SEQ * DINNER;

  for (int l = 0; l < 2; ++l) {
    float* hp = (l == 0) ? h   : hflip;
    float* rp = (l == 0) ? res : resflip;
    if (l == 1) {   // bidirectional: flip h and res, then res = h + res
      flip_kernel<<<ceil_div((int)LD, 256), 256, 0, stream>>>(h, hflip);
      flip_kernel<<<ceil_div((int)LD, 256), 256, 0, stream>>>(res, resflip);
      add_kernel<<<ceil_div((int)LD, 256), 256, 0, stream>>>(hflip, resflip, resflip, (int)LD);
    }
    // mamba mixer
    rmsnorm_bf_kernel<<<L_SEQ, 256, 0, stream>>>(rp, Wp[l][9], xnorm_bf);
    gemm(xnorm_bf, DMODEL, wt[l][0], nullptr, xz, nullptr, 2 * DINNER, DMODEL, 0);    // in_proj
    conv_silu_kernel<<<ceil_div(nEl, 256), 256, 0, stream>>>(xz, Wp[l][1], Wp[l][2],
                                                             xconv, xconv_bf, nEl);
    gemm(xconv_bf, DINNER, wt[l][1], nullptr, dbc, dbc_bf, NXPROJ, DINNER, 0);        // x_proj
    gemm(dbc_bf, NXPROJ, wt[l][2], Wp[l][5], dtb, nullptr, DINNER, DTRANK, 2);        // dt + softplus
    scan_kernel<<<DINNER / 8, 256, 0, stream>>>(xconv, dtb, Wp[l][6], dbc, Wp[l][7], yscan);
    gate_mul_kernel<<<ceil_div(nEl, 256), 256, 0, stream>>>(yscan, xz, ygated_bf, nEl);
    gemm(ygated_bf, DINNER, wt[l][3], nullptr, mout, nullptr, DMODEL, DINNER, 0);     // out_proj
    add_kernel<<<ceil_div((int)LD, 256), 256, 0, stream>>>(mout, rp, rp, (int)LD);    // res += mixer
    // gated MLP
    rmsnorm_bf_kernel<<<L_SEQ, 256, 0, stream>>>(rp, Wp[l][10], h2n_bf);
    gemm(h2n_bf, DMODEL, wt[l][4], Wp[l][12], fcout, nullptr, 2 * NHID, DMODEL, 1);   // fc1 + bias
    glu_kernel<<<ceil_div(nEl, 256), 256, 0, stream>>>(fcout, glub_bf, nEl);
    gemm(glub_bf, NHID, wt[l][5], Wp[l][14], hp, nullptr, DMODEL, NHID, 1);           // fc2 + bias
    // capture ntp / ptp = h + res
    add_kernel<<<ceil_div((int)LD, 256), 256, 0, stream>>>(hp, rp, (l == 0) ? ntp : ptp, (int)LD);
  }

  float* out = (float*)d_out;
  layernorm_kernel<<<L_SEQ, 256, 0, stream>>>(ntp, lnw, lnb, out);         // ln(ntp)
  layernorm_kernel<<<L_SEQ, 256, 0, stream>>>(ptp, lnw, lnb, out + LD);    // ln(ptp)
}
